// VectorQuantizerEMA_24369644437759
// MI455X (gfx1250) — matrix operand order repaired and checked by reading, not compiled, not run on hardware
//
#include <hip/hip_runtime.h>
#include <hip/hip_bf16.h>
#include <math.h>

typedef __attribute__((ext_vector_type(16))) _Float16 v16h;
typedef __attribute__((ext_vector_type(8)))  _Float16 v8h;
typedef __attribute__((ext_vector_type(8)))  float    v8f;

#define NUM_CODES 8192
#define CODE_DIM  512
#define BATCH     32768
#define DECAY     0.995f
#define EPS       1e-8f
#define COMMIT    0.3f
#define DIVW      0.001f

// ---------------- output layout (floats, return order) ----------------
#define O_ZQ   0                       // [32768*512]
#define O_VQ   16777216                // [1]
#define O_PERP 16777217                // [1]
#define O_IDX  16777218                // [32768] (as float)
#define O_DIV  16809986                // [1]
#define O_NEMB 16809987                // [8192*512]
#define O_NCS  21004291                // [8192]
#define O_FEA  21012483                // [8192*512]

// ---------------- workspace layout (bytes) ----------------
#define WSB_ZN   0                       // f16 [32768*512] -> 33554432 B
#define WSB_EN   33554432u               // f16 [8192*512]  ->  8388608 B
#define WSB_ESUM 41943040u               // f32 [8192*512]  -> 16777216 B
#define WSB_CNT  58720256u               // f32 [8192]      ->    32768 B
#define WSB_SCAL 58753024u               // f32 [16] (n, loss, pad)
#define WSB_IDX  58753088u               // i32 [32768]
#define N_ZERO_F ((16777216u + 32768u + 64u) / 4u)   // esum + counts + scalars

__global__ __launch_bounds__(256) void vq_zero(float* __restrict__ p, unsigned n) {
    unsigned i = blockIdx.x * 256u + threadIdx.x;
    if (i < n) p[i] = 0.0f;
}

// one block per row: L2-normalize a row into f16
__global__ __launch_bounds__(256) void vq_norm_rows(const float* __restrict__ in,
                                                    _Float16* __restrict__ out) {
    __shared__ float red[256];
    const int row = blockIdx.x;
    const float* r = in + (size_t)row * CODE_DIM;
    float s = 0.0f;
    for (int i = threadIdx.x; i < CODE_DIM; i += 256) { float v = r[i]; s += v * v; }
    red[threadIdx.x] = s; __syncthreads();
    for (int o = 128; o > 0; o >>= 1) {
        if ((int)threadIdx.x < o) red[threadIdx.x] += red[threadIdx.x + o];
        __syncthreads();
    }
    const float inv = 1.0f / (sqrtf(red[0]) + EPS);
    _Float16* w = out + (size_t)row * CODE_DIM;
    for (int i = threadIdx.x; i < CODE_DIM; i += 256) w[i] = (_Float16)(r[i] * inv);
}

// B 32x16 layout fetch: lane = N (code), per-lane low 8 = K [half*8, +8), high 8 = K [16+half*8, +8).
// bbase = en + col*CODE_DIM + half*8 ; element offset = nt*(16*CODE_DIM) + t*32
__device__ __forceinline__ v16h vq_load_b(const _Float16* __restrict__ bbase,
                                          int nt, int t) {
    const _Float16* p = bbase + (size_t)nt * (16 * CODE_DIM) + t * 32;
    const v8h lo = *(const v8h*)p;
    const v8h hi = *(const v8h*)(p + 16);
    return __builtin_shufflevector(lo, hi, 0,1,2,3,4,5,6,7,8,9,10,11,12,13,14,15);
}

// ---------- WMMA similarity + per-row argmax ----------
// 8 waves/block; each wave owns 32 rows of z (two 16-row A register tiles), so
// every streamed B tile feeds TWO v_wmma issues (1 b128 load : 1 wmma).
// B is explicitly double-buffered (Bcur/Bnext) so the next load pair is in
// flight while the current WMMA pair executes.
__global__ __launch_bounds__(256) void vq_sim_argmax(const _Float16* __restrict__ zn,
                                                     const _Float16* __restrict__ en,
                                                     int* __restrict__ idx_out,
                                                     float* __restrict__ idx_out_f) {
    const int lane = threadIdx.x & 31;
    const int wave = threadIdx.x >> 5;
    const int row_base = (blockIdx.x * 8 + wave) * 32;
    const int col  = lane & 15;   // N within tile / M row selector for A
    const int half = lane >> 4;   // lane half

    // 16-bit A layout: per-lane low 8 halves = K [half*8, +8), high 8 = K [16+half*8, +8)
    const _Float16* arow0 = zn + (size_t)(row_base + col) * CODE_DIM;
    const _Float16* arow1 = zn + (size_t)(row_base + 16 + col) * CODE_DIM;
    v16h A0[16], A1[16];
#pragma unroll
    for (int t = 0; t < 16; ++t) {
        const v8h lo0 = *(const v8h*)(arow0 + t * 32 + half * 8);
        const v8h hi0 = *(const v8h*)(arow0 + t * 32 + 16 + half * 8);
        A0[t] = __builtin_shufflevector(lo0, hi0, 0,1,2,3,4,5,6,7,8,9,10,11,12,13,14,15);
        const v8h lo1 = *(const v8h*)(arow1 + t * 32 + half * 8);
        const v8h hi1 = *(const v8h*)(arow1 + t * 32 + 16 + half * 8);
        A1[t] = __builtin_shufflevector(lo1, hi1, 0,1,2,3,4,5,6,7,8,9,10,11,12,13,14,15);
    }

    float best0[8], best1[8]; int bidx0[8], bidx1[8];
#pragma unroll
    for (int r = 0; r < 8; ++r) {
        best0[r] = -3.402823e38f; bidx0[r] = 0;
        best1[r] = -3.402823e38f; bidx1[r] = 0;
    }

    const _Float16* bbase = en + (size_t)col * CODE_DIM + half * 8;
    v16h Bcur = vq_load_b(bbase, 0, 0);   // prime the pipeline

    for (int nt = 0; nt < NUM_CODES / 16; ++nt) {
        const int code = nt * 16 + col;
        v8f acc0 = {}, acc1 = {};
#pragma unroll
        for (int t = 0; t < 16; ++t) {
            // prefetch the next sub-tile (wraps into next N-tile; 511->0 harmless)
            const int nnt = (t == 15) ? ((nt + 1) & (NUM_CODES / 16 - 1)) : nt;
            const int ntt = (t == 15) ? 0 : (t + 1);
            const v16h Bnext = vq_load_b(bbase, nnt, ntt);
            acc0 = __builtin_amdgcn_wmma_f32_16x16x32_f16(false, A0[t], false, Bcur,
                                                          (short)0, acc0, false, false);
            acc1 = __builtin_amdgcn_wmma_f32_16x16x32_f16(false, A1[t], false, Bcur,
                                                          (short)0, acc1, false, false);
            Bcur = Bnext;
        }
#pragma unroll
        for (int r = 0; r < 8; ++r) {
            const float v0 = acc0[r];
            if (v0 > best0[r]) { best0[r] = v0; bidx0[r] = code; }  // '>' keeps first max
            const float v1 = acc1[r];
            if (v1 > best1[r]) { best1[r] = v1; bidx1[r] = code; }
        }
    }

    // reduce across the 16 lanes of each half (butterfly stays within a half)
#pragma unroll
    for (int r = 0; r < 8; ++r) {
#pragma unroll
        for (int m = 8; m >= 1; m >>= 1) {
            const float ov0 = __shfl_xor(best0[r], m, 32);
            const int   oi0 = __shfl_xor(bidx0[r], m, 32);
            if (ov0 > best0[r] || (ov0 == best0[r] && oi0 < bidx0[r])) { best0[r] = ov0; bidx0[r] = oi0; }
            const float ov1 = __shfl_xor(best1[r], m, 32);
            const int   oi1 = __shfl_xor(bidx1[r], m, 32);
            if (ov1 > best1[r] || (ov1 == best1[r] && oi1 < bidx1[r])) { best1[r] = ov1; bidx1[r] = oi1; }
        }
        if (col == 0) {
            const int row0 = row_base + r + 8 * half;        // C slot r -> M = r + 8*half
            idx_out[row0]   = bidx0[r];
            idx_out_f[row0] = (float)bidx0[r];
            const int row1 = row_base + 16 + r + 8 * half;
            idx_out[row1]   = bidx1[r];
            idx_out_f[row1] = (float)bidx1[r];
        }
    }
}

// gather z_q (== z_q_st), segment-sum z into embed_sum, counts, squared-error accum
__global__ __launch_bounds__(256) void vq_gather_update(const float* __restrict__ z,
                                                        const float* __restrict__ emb,
                                                        const int* __restrict__ idx,
                                                        float* __restrict__ zq_out,
                                                        float* __restrict__ counts,
                                                        float* __restrict__ embed_sum,
                                                        float* __restrict__ loss_accum) {
    __shared__ float red[256];
    const int row = blockIdx.x;
    const int k = idx[row];
    const float* zr = z   + (size_t)row * CODE_DIM;
    const float* er = emb + (size_t)k   * CODE_DIM;
    float*       o  = zq_out + (size_t)row * CODE_DIM;
    float*       es = embed_sum + (size_t)k * CODE_DIM;
    float s = 0.0f;
    for (int i = threadIdx.x; i < CODE_DIM; i += 256) {
        const float zv = zr[i], ev = er[i];
        o[i] = ev;                       // z + sg(z_q - z) == z_q
        const float d = zv - ev;
        s += d * d;
        atomicAdd(&es[i], zv);
    }
    red[threadIdx.x] = s; __syncthreads();
    for (int off = 128; off > 0; off >>= 1) {
        if ((int)threadIdx.x < off) red[threadIdx.x] += red[threadIdx.x + off];
        __syncthreads();
    }
    if (threadIdx.x == 0) {
        atomicAdd(loss_accum, red[0]);
        atomicAdd(&counts[k], 1.0f);
    }
}

// new_cluster_size = decay*cs + (1-decay)*counts ; accumulate n = sum(ncs)
__global__ __launch_bounds__(256) void vq_cluster_update(const float* __restrict__ cs,
                                                         const float* __restrict__ counts,
                                                         float* __restrict__ ncs_out,
                                                         float* __restrict__ n_accum) {
    __shared__ float red[256];
    const int i = blockIdx.x * 256 + threadIdx.x;
    const float v = DECAY * cs[i] + (1.0f - DECAY) * counts[i];
    ncs_out[i] = v;
    red[threadIdx.x] = v; __syncthreads();
    for (int off = 128; off > 0; off >>= 1) {
        if ((int)threadIdx.x < off) red[threadIdx.x] += red[threadIdx.x + off];
        __syncthreads();
    }
    if (threadIdx.x == 0) atomicAdd(n_accum, red[0]);
}

// per-code: embed_avg EMA, laplace-smooth, re-normalize, final_embed_avg
__global__ __launch_bounds__(256) void vq_embed_update(const float* __restrict__ embed_avg,
                                                       const float* __restrict__ embed_sum,
                                                       const float* __restrict__ ncs,
                                                       const float* __restrict__ n_ptr,
                                                       float* __restrict__ new_emb,
                                                       float* __restrict__ final_ea) {
    __shared__ float red[256];
    __shared__ float buf[CODE_DIM];
    const int k = blockIdx.x;
    const float n = *n_ptr;
    const float csn = (ncs[k] + EPS) / (n + (float)NUM_CODES * EPS) * n;
    const float inv_csn = 1.0f / csn;
    const float* ea = embed_avg + (size_t)k * CODE_DIM;
    const float* es = embed_sum + (size_t)k * CODE_DIM;
    float s = 0.0f;
    for (int i = threadIdx.x; i < CODE_DIM; i += 256) {
        const float u = (DECAY * ea[i] + (1.0f - DECAY) * es[i]) * inv_csn;
        buf[i] = u; s += u * u;
    }
    red[threadIdx.x] = s; __syncthreads();
    for (int off = 128; off > 0; off >>= 1) {
        if ((int)threadIdx.x < off) red[threadIdx.x] += red[threadIdx.x + off];
        __syncthreads();
    }
    const float inv = 1.0f / (sqrtf(red[0]) + EPS);
    float* ne = new_emb  + (size_t)k * CODE_DIM;
    float* fe = final_ea + (size_t)k * CODE_DIM;
    for (int i = threadIdx.x; i < CODE_DIM; i += 256) {
        const float v = buf[i] * inv;
        ne[i] = v;
        fe[i] = v * csn;
    }
}

// scalars: entropy/diversity/perplexity/vq_loss
__global__ __launch_bounds__(256) void vq_scalars(const float* __restrict__ counts,
                                                  const float* __restrict__ loss_ptr,
                                                  float* __restrict__ out_vq,
                                                  float* __restrict__ out_perp,
                                                  float* __restrict__ out_div) {
    __shared__ float red[256];
    __shared__ float totalS;
    float s = 0.0f;
    for (int i = threadIdx.x; i < NUM_CODES; i += 256) s += counts[i];
    red[threadIdx.x] = s; __syncthreads();
    for (int off = 128; off > 0; off >>= 1) {
        if ((int)threadIdx.x < off) red[threadIdx.x] += red[threadIdx.x + off];
        __syncthreads();
    }
    if (threadIdx.x == 0) totalS = red[0];
    __syncthreads();
    const float inv_total = 1.0f / totalS;
    const float inv_b = 1.0f / (float)BATCH;
    float e = 0.0f, eb = 0.0f;
    for (int i = threadIdx.x; i < NUM_CODES; i += 256) {
        const float c = counts[i];
        const float p  = c * inv_total;
        const float ap = c * inv_b;
        e  -= p  * logf(p  + 1e-10f);
        eb -= ap * logf(ap + 1e-10f);
    }
    red[threadIdx.x] = e; __syncthreads();
    for (int off = 128; off > 0; off >>= 1) {
        if ((int)threadIdx.x < off) red[threadIdx.x] += red[threadIdx.x + off];
        __syncthreads();
    }
    const float entropy = red[0];
    __syncthreads();
    red[threadIdx.x] = eb; __syncthreads();
    for (int off = 128; off > 0; off >>= 1) {
        if ((int)threadIdx.x < off) red[threadIdx.x] += red[threadIdx.x + off];
        __syncthreads();
    }
    if (threadIdx.x == 0) {
        const float entropy_b = red[0];
        const float max_e = logf((float)NUM_CODES);
        const float div = (max_e - entropy) / max_e;
        const float mse = (*loss_ptr) / ((float)BATCH * (float)CODE_DIM);
        *out_div  = div;
        *out_perp = expf(entropy_b);
        *out_vq   = mse * COMMIT + mse + DIVW * div;
    }
}

extern "C" void kernel_launch(void* const* d_in, const int* in_sizes, int n_in,
                              void* d_out, int out_size, void* d_ws, size_t ws_size,
                              hipStream_t stream) {
    const float* z    = (const float*)d_in[0];
    const float* emb  = (const float*)d_in[1];
    const float* cs   = (const float*)d_in[2];
    const float* eavg = (const float*)d_in[3];
    float* out = (float*)d_out;
    char*  ws  = (char*)d_ws;

    _Float16* zn_h  = (_Float16*)(ws + WSB_ZN);
    _Float16* en_h  = (_Float16*)(ws + WSB_EN);
    float*    esum  = (float*)(ws + WSB_ESUM);
    float*    cnts  = (float*)(ws + WSB_CNT);
    float*    scal  = (float*)(ws + WSB_SCAL);   // [0]=n_accum, [1]=loss_accum
    int*      idx   = (int*)(ws + WSB_IDX);

    // 1) zero accumulators (esum + counts + scalars, contiguous)
    vq_zero<<<(N_ZERO_F + 255u) / 256u, 256, 0, stream>>>(esum, N_ZERO_F);

    // 2) normalize rows of z and embedding into f16
    vq_norm_rows<<<BATCH, 256, 0, stream>>>(z, zn_h);
    vq_norm_rows<<<NUM_CODES, 256, 0, stream>>>(emb, en_h);

    // 3) WMMA similarity + argmax (1024 waves, 32 rows each, 8 waves/block)
    vq_sim_argmax<<<BATCH / 32 / 8, 256, 0, stream>>>(zn_h, en_h, idx, out + O_IDX);

    // 4) gather z_q, segment sums, loss accum
    vq_gather_update<<<BATCH, 256, 0, stream>>>(z, emb, idx, out + O_ZQ, cnts, esum,
                                                scal + 1);

    // 5) cluster size EMA + n
    vq_cluster_update<<<NUM_CODES / 256, 256, 0, stream>>>(cs, cnts, out + O_NCS, scal);

    // 6) embedding EMA + renorm + final_embed_avg
    vq_embed_update<<<NUM_CODES, 256, 0, stream>>>(eavg, esum, out + O_NCS, scal,
                                                   out + O_NEMB, out + O_FEA);

    // 7) scalar losses
    vq_scalars<<<1, 256, 0, stream>>>(cnts, scal + 1, out + O_VQ, out + O_PERP,
                                      out + O_DIV);
}
